// Encoder_68324339745355
// MI455X (gfx1250) — matrix operand order repaired and checked by reading, compiled, not run on hardware
//
#include <hip/hip_runtime.h>

typedef __attribute__((ext_vector_type(16))) _Float16 v16h;
typedef __attribute__((ext_vector_type(8)))  _Float16 v8h;
typedef __attribute__((ext_vector_type(8)))  float    v8f;
typedef __attribute__((ext_vector_type(4)))  float    v4f;

#define TT 32
#define BB 16384
#define HH 64
#define EE 64
#define GG 256      // 4*H gate width
#define KC 128      // E + H concatenated K
#define ROWS 32     // batch rows per block

__launch_bounds__(256, 1)
__global__ void slstm_fused_v2(const float* __restrict__ obs,
                               const float* __restrict__ img1,
                               const float* __restrict__ img2,
                               const float* __restrict__ Wsp,
                               const float* __restrict__ bsp,
                               const float* __restrict__ Wih,
                               const float* __restrict__ Whh,
                               const float* __restrict__ bg,
                               float* __restrict__ out)
{
    __shared__ _Float16 sX[ROWS][KC];   // [row][k]: k<64 = x_t, k>=64 = h_{t-1}
    __shared__ v4f      sWsp4[EE];      // W_sp rows (E x 4) as float4
    __shared__ float    sBgF[GG];       // folded bias: b_g + W_ih @ b_sp

    const int tid   = threadIdx.x;
    const int lane  = tid & 31;
    const int wave  = tid >> 5;
    const int rtile = wave >> 2;        // 0/1 : rows [16*rtile, 16*rtile+16)
    const int u     = wave & 3;         // gate e-group: e in [16u, 16u+16)
    const int rowBase = blockIdx.x * ROWS;

    // ---- one-time LDS fills ----
    if (tid < EE) sWsp4[tid] = ((const v4f*)Wsp)[tid];
    {   // fold b_sp through W_ih into the gate bias (exact algebra)
        float acc = bg[tid];
        const float* wr = Wih + tid * EE;
        #pragma unroll 8
        for (int e = 0; e < EE; ++e) acc += bsp[e] * wr[e];
        sBgF[tid] = acc;
    }
    // x-compute ownership: this thread owns row xr, cols [e0, e0+8)
    const int xr = tid >> 3;
    const int e0 = (tid & 7) * 8;
    {   // zero h columns (h0 = 0): one b128 store per thread
        v8h z = {};
        *(v8h*)&sX[xr][64 + e0] = z;
    }

    // ---- B fragments in registers: Wcat[N][K=128] = [W_ih | W_hh] ----
    // wave owns gate tiles {u, u+4, u+8, u+12} (i, f, z, o); 4 K-frags each.
    // B layout used here (contiguous K, same order as A): lane holds column N = tile*16 + lane%16, 16 consecutive K;
    // lanes 0-15 -> K base+0..15, lanes 16-31 -> K base+16..31.
    v16h bfr[4][4];
    {
        const int col16 = lane & 15;
        const int khalf = (lane < 16) ? 0 : 16;
        #pragma unroll
        for (int g = 0; g < 4; ++g) {
            int c = g * 64 + u * 16 + col16;           // gate row 0..255
            #pragma unroll
            for (int kf = 0; kf < 4; ++kf) {
                int kcat = kf * 32 + khalf;            // 16 contiguous K
                const float* src = (kcat < 64) ? (Wih + c * 64 + kcat)
                                               : (Whh + c * 64 + (kcat - 64));
                v16h f;
                #pragma unroll
                for (int j = 0; j < 16; ++j) f[j] = (_Float16)src[j];
                bfr[g][kf] = f;
            }
        }
    }
    __syncthreads();

    // per-lane gate biases: e is lane-invariant across accumulator rows
    const int eg  = u * 16 + (lane & 15);
    const float bgi = sBgF[eg];
    const float bgf = sBgF[64 + eg];
    const float bgz = sBgF[128 + eg];
    const float bgo = sBgF[192 + eg];

    // state in C/D layout: lane j-th element <-> (row 16*rtile + mofs + j, col eg)
    float cs[8] = {0.f,0.f,0.f,0.f,0.f,0.f,0.f,0.f};
    float ns[8] = {0.f,0.f,0.f,0.f,0.f,0.f,0.f,0.f};

    const size_t xrow = (size_t)rowBase + xr;

    // register prefetch (t = 0)
    v4f p1a, p1b, p2a, p2b, pob;
    {
        const float* b1 = img1 + xrow * EE + e0;
        const float* b2 = img2 + xrow * EE + e0;
        p1a = __builtin_nontemporal_load((const v4f*)(b1));
        p1b = __builtin_nontemporal_load((const v4f*)(b1 + 4));
        p2a = __builtin_nontemporal_load((const v4f*)(b2));
        p2b = __builtin_nontemporal_load((const v4f*)(b2 + 4));
        pob = *(const v4f*)(obs + xrow * 4);
    }

    const float L2E = 1.4426950408889634f;
    const int   ar   = rtile * 16 + (lane & 15);   // A-frag row for this lane
    const int   koff = (lane < 16) ? 0 : 16;
    const int   mofs = (lane < 16) ? 0 : 8;

    for (int t = 0; t < TT; ++t) {
        // ---- 1) x_t = obs@Wsp^T + img1 + img2 -> sX cols [e0,e0+8) as f16 ----
        {
            v8h xh;
            #pragma unroll
            for (int j = 0; j < 8; ++j) {
                float pj1 = (j < 4) ? p1a[j] : p1b[j - 4];
                float pj2 = (j < 4) ? p2a[j] : p2b[j - 4];
                v4f w = sWsp4[e0 + j];
                float x = pj1 + pj2 + pob[0] * w[0] + pob[1] * w[1]
                                    + pob[2] * w[2] + pob[3] * w[3];
                xh[j] = (_Float16)x;
            }
            *(v8h*)&sX[xr][e0] = xh;   // single ds_store_b128
        }
        __syncthreads();   // b1: x_t and h_{t-1} visible to all waves

        // ---- 2) prefetch t+1 (hidden behind WMMA) ----
        if (t + 1 < TT) {
            size_t base = ((size_t)(t + 1) * BB + xrow);
            const float* b1 = img1 + base * EE + e0;
            const float* b2 = img2 + base * EE + e0;
            p1a = __builtin_nontemporal_load((const v4f*)(b1));
            p1b = __builtin_nontemporal_load((const v4f*)(b1 + 4));
            p2a = __builtin_nontemporal_load((const v4f*)(b2));
            p2b = __builtin_nontemporal_load((const v4f*)(b2 + 4));
            pob = *(const v4f*)(obs + base * 4);
        }

        // ---- 3) g = [x|h] @ Wcat^T : 4 accumulators (i,f,z,o), K=128 ----
        v8f acc0 = {}, acc1 = {}, acc2 = {}, acc3 = {};
        #pragma unroll
        for (int kf = 0; kf < 4; ++kf) {
            v8h lo = *(const v8h*)&sX[ar][kf * 32 + koff];
            v8h hi = *(const v8h*)&sX[ar][kf * 32 + 8 + koff];
            v16h a;
            #pragma unroll
            for (int j = 0; j < 8; ++j) { a[j] = lo[j]; a[8 + j] = hi[j]; }
            acc0 = __builtin_amdgcn_wmma_f32_16x16x32_f16(
                       false, a, false, bfr[0][kf], (short)0, acc0, false, false);
            acc1 = __builtin_amdgcn_wmma_f32_16x16x32_f16(
                       false, a, false, bfr[1][kf], (short)0, acc1, false, false);
            acc2 = __builtin_amdgcn_wmma_f32_16x16x32_f16(
                       false, a, false, bfr[2][kf], (short)0, acc2, false, false);
            acc3 = __builtin_amdgcn_wmma_f32_16x16x32_f16(
                       false, a, false, bfr[3][kf], (short)0, acc3, false, false);
        }
        __syncthreads();   // b2: all A-frag reads done before h is overwritten

        // ---- 4) gates directly on C/D-layout accumulators ----
        #pragma unroll
        for (int j = 0; j < 8; ++j) {
            int row = rtile * 16 + mofs + j;       // block-local batch row
            float it = acc0[j] + bgi;
            float ft = acc1[j] + bgf;
            float zt = acc2[j] + bgz;
            float ot = acc3[j] + bgo;

            float i  = __builtin_amdgcn_exp2f(it * L2E);
            float fg = __builtin_amdgcn_rcpf(1.0f + __builtin_amdgcn_exp2f(-ft * L2E));
            float e2 = __builtin_amdgcn_exp2f(zt * (2.0f * L2E));
            float z  = 1.0f - 2.0f * __builtin_amdgcn_rcpf(e2 + 1.0f);
            float og = __builtin_amdgcn_rcpf(1.0f + __builtin_amdgcn_exp2f(-ot * L2E));

            float c = fg * cs[j] + i * z;
            float n = fg * ns[j] + i;
            cs[j] = c; ns[j] = n;
            float h = og * c * __builtin_amdgcn_rcpf(n);

            sX[row][64 + eg] = (_Float16)h;        // h_t for next step's A frag

            size_t brow = (size_t)rowBase + row;
            __builtin_nontemporal_store(h, &out[((size_t)t * BB + brow) * HH + eg]);
            if (t == TT - 1) {
                size_t base = (size_t)TT * BB * HH;
                out[base + brow * HH + eg]                       = h;  // hT
                out[base + (size_t)BB * HH + brow * HH + eg]     = c;  // cT
                out[base + 2 * (size_t)BB * HH + brow * HH + eg] = n;  // nT
            }
        }
        // loop-top barrier (b1 of t+1) fences these h writes from next A reads
    }
}

extern "C" void kernel_launch(void* const* d_in, const int* in_sizes, int n_in,
                              void* d_out, int out_size, void* d_ws, size_t ws_size,
                              hipStream_t stream) {
    const float* obs  = (const float*)d_in[0];
    const float* img1 = (const float*)d_in[1];
    const float* img2 = (const float*)d_in[2];
    const float* Wsp  = (const float*)d_in[3];
    const float* bsp  = (const float*)d_in[4];
    const float* Wih  = (const float*)d_in[5];
    const float* Whh  = (const float*)d_in[6];
    const float* bg   = (const float*)d_in[7];
    float* out = (float*)d_out;

    dim3 grid(BB / ROWS), block(256);
    hipLaunchKernelGGL(slstm_fused_v2, grid, block, 0, stream,
                       obs, img1, img2, Wsp, bsp, Wih, Whh, bg, out);
}
